// KernelAttentionRegression_54743653154995
// MI455X (gfx1250) — matrix operand order repaired and checked by reading, not compiled, not run on hardware
//
#include <hip/hip_runtime.h>
#include <hip/hip_bf16.h>
#include <math.h>

// ---------------------------------------------------------------------------
// Kernel attention regression on MI455X (gfx1250, wave32).
// Heavy matmuls run as bf16 WMMA (v_wmma_f32_16x16x32_bf16, f32 accum).
// Global->LDS staging uses CDNA5 async-LDS loads (ASYNCcnt) with double
// buffering so DMA of tile t+1 overlaps WMMA on tile t.
// K matrix kept bf16 (256MB); per-(b,h) 8MB slices stay L2-resident.
// ---------------------------------------------------------------------------

typedef __bf16 bf16;
typedef __attribute__((ext_vector_type(16))) __bf16 v16bf;
typedef __attribute__((ext_vector_type(8)))  __bf16 v8bf;
typedef __attribute__((ext_vector_type(8)))  float  v8f;
typedef __attribute__((ext_vector_type(4)))  int    v4i;

#define S_LEN   2048
#define D_DIM   1024
#define HEADS   16
#define HDIM    64
#define RANK    16
#define BATCH   2
#define NZ      (BATCH * HEADS)          // 32 (b,h) pairs
#define N_ITERS 8
#define CLIP_ABS 10.0f
#define EPS_F    1e-6f

// -------------------- CDNA5 async global->LDS path -------------------------
#if defined(__has_builtin)
#  if __has_builtin(__builtin_amdgcn_global_load_async_to_lds_b128)
#    define USE_ASYNC_LDS 1
#  endif
#endif
#ifndef USE_ASYNC_LDS
#  define USE_ASYNC_LDS 0
#endif

__device__ __forceinline__ void copy16(const bf16* __restrict__ g, bf16* l) {
#if USE_ASYNC_LDS
  // builtin expects: (int4 __device__* src, int4 __shared__* dst, imm offset, imm cpol)
  __builtin_amdgcn_global_load_async_to_lds_b128(
      (__attribute__((address_space(1))) v4i*)(void*)g,
      (__attribute__((address_space(3))) v4i*)(void*)l, 0, 0);
#else
  *(v8bf*)l = *(const v8bf*)g;
#endif
}

template <int N>
__device__ __forceinline__ void wait_asynccnt() {
#if USE_ASYNC_LDS
#  if __has_builtin(__builtin_amdgcn_s_wait_asynccnt)
  __builtin_amdgcn_s_wait_asynccnt(N);
#  else
  asm volatile("s_wait_asynccnt %0" ::"i"(N) : "memory");
#  endif
#endif
}

union FragBF { v16bf v; v8bf h[2]; };

__device__ __forceinline__ float softplusf(float x) {
  return (x > 20.0f) ? x : log1pf(expf(x));
}

// ---------------------------------------------------------------------------
// Elementwise / small kernels
// ---------------------------------------------------------------------------

__global__ void cast_f32_to_bf16_k(const float* __restrict__ in,
                                   bf16* __restrict__ out, long n) {
  long i = (long)blockIdx.x * blockDim.x + threadIdx.x;
  if (i < n) out[i] = (bf16)in[i];
}

__global__ void sqnorm64_k(const bf16* __restrict__ q, float* __restrict__ out,
                           int rows) {
  int i = blockIdx.x * blockDim.x + threadIdx.x;
  if (i >= rows) return;
  const bf16* p = q + (size_t)i * HDIM;
  float s = 0.f;
#pragma unroll
  for (int d = 0; d < HDIM; ++d) { float v = (float)p[d]; s += v * v; }
  out[i] = s;
}

__global__ void diag_precond_k(const bf16* __restrict__ q, const bf16* __restrict__ k,
                               const float* __restrict__ bandwidth,
                               const float* __restrict__ diag_scale,
                               const float* __restrict__ reg,
                               float* __restrict__ dp) {
  int i = blockIdx.x * blockDim.x + threadIdx.x;      // z*S + s
  if (i >= NZ * S_LEN) return;
  int z = i >> 11;
  int h = z & (HEADS - 1);
  const bf16* qp = q + (size_t)i * HDIM;
  const bf16* kp = k + (size_t)i * HDIM;
  float ss = 0.f;
#pragma unroll
  for (int d = 0; d < HDIM; ++d) {
    float dl = (float)qp[d] - (float)kp[d];
    ss += dl * dl;
  }
  float bw = softplusf(*bandwidth) + EPS_F;
  float kd = expf(-ss / (2.f * bw * bw));
  dp[i] = softplusf(kd) * diag_scale[h] + *reg;
}

__global__ void lowrank_k(const float* __restrict__ pos, const float* __restrict__ hp,
                          float* __restrict__ lr) {
  int i = blockIdx.x * blockDim.x + threadIdx.x;      // h*S*R + s*R + r
  if (i >= HEADS * S_LEN * RANK) return;
  int h = i >> 15;
  int rem = i & (S_LEN * RANK - 1);
  int s = rem >> 4;
  int r = rem & (RANK - 1);
  float acc = 0.f;
#pragma unroll
  for (int j = 0; j < RANK; ++j)
    acc += pos[s * RANK + j] * hp[(h * RANK + j) * RANK + r];
  lr[i] = acc;
}

// t[z][r][d] = sum_s lr[h][s][r] * res[z][s][d]
__global__ void rich_reduce_k(const float* __restrict__ lr,
                              const float* __restrict__ res,
                              float* __restrict__ t) {
  int z = blockIdx.x;
  int tid = threadIdx.x;            // 0..1023
  int r = tid >> 6;
  int d = tid & (HDIM - 1);
  int h = z & (HEADS - 1);
  const float* lrh = lr + (size_t)h * S_LEN * RANK;
  const float* rz  = res + (size_t)z * S_LEN * HDIM;
  float acc = 0.f;
  for (int s = 0; s < S_LEN; ++s)
    acc += lrh[s * RANK + r] * rz[s * HDIM + d];
  t[((size_t)z * RANK + r) * HDIM + d] = acc;
}

__global__ void rich_update_k(const float* __restrict__ res,
                              const float* __restrict__ diagp,
                              const float* __restrict__ lr,
                              const float* __restrict__ t,
                              float* __restrict__ alpha,
                              bf16* __restrict__ alphaT) {
  long i = (long)blockIdx.x * blockDim.x + threadIdx.x;   // z*S*HD
  if (i >= (long)NZ * S_LEN * HDIM) return;
  int z = (int)(i >> 17);
  int rem = (int)(i & (S_LEN * HDIM - 1));
  int s = rem >> 6;
  int d = rem & (HDIM - 1);
  int h = z & (HEADS - 1);
  float pre = res[i] * diagp[z * S_LEN + s];
  const float* lrs = lr + ((size_t)h * S_LEN + s) * RANK;
  const float* tz  = t + ((size_t)z * RANK) * HDIM + d;
#pragma unroll
  for (int r = 0; r < RANK; ++r) pre += lrs[r] * tz[r * HDIM];
  float a = alpha[i] + pre;
  a = fminf(fmaxf(a, -CLIP_ABS), CLIP_ABS);
  alpha[i] = a;
  alphaT[((size_t)z * HDIM + d) * S_LEN + s] = (bf16)a;
}

// ---------------------------------------------------------------------------
// Generic NT bf16 WMMA GEMM: C[M,N] = A[M,K] * B[N,K]^T  (both K-contiguous)
// Async double-buffered LDS staging. Epilogue modes:
//  0: bf16 head-split [b,h,s,hd]   1: RBF exp -> bf16 K (z-strided)
//  2: residual = v-(c+lam*a) f32   3: bf16 attn interleave   4: plain f32
// ---------------------------------------------------------------------------
template <int BM, int BN, int WR, int WC>
__global__ __launch_bounds__(WR * WC * 32)
void wmma_gemm_nt(const bf16* __restrict__ A, const bf16* __restrict__ B,
                  int M, int N, int K, long aStrideZ, long bStrideZ,
                  int mode, void* __restrict__ out, long outStrideZBytes,
                  const void* __restrict__ auxp0, const void* __restrict__ auxp1,
                  long aux0StrideZ, long aux1StrideZ,
                  const float* __restrict__ scal0) {
  constexpr int KT = 32;
  constexpr int LDSW = KT + 8;                 // pad to dodge bank conflicts
  constexpr int NTHR = WR * WC * 32;
  constexpr int TM = BM / (WR * 16);
  constexpr int TN = BN / (WC * 16);
  constexpr int ROWS_PER_PASS = NTHR / 4;      // 4 x 16B chunks per 32-elem row
  constexpr int ACH = BM / ROWS_PER_PASS;      // A chunks per thread per tile
  constexpr int BCH = BN / ROWS_PER_PASS;      // B chunks per thread per tile
  constexpr int CHUNKS = ACH + BCH;

  __shared__ bf16 lA[2][BM][LDSW];
  __shared__ bf16 lB[2][BN][LDSW];

  const int z  = blockIdx.z;
  const int m0 = blockIdx.y * BM;
  const int n0 = blockIdx.x * BN;
  const int tid  = threadIdx.x;
  const int wave = tid >> 5;
  const int lane = tid & 31;
  const int wr = wave / WC, wc = wave % WC;
  const int wmBase = wr * TM * 16;
  const int wnBase = wc * TN * 16;

  const bf16* Az = A + (size_t)z * aStrideZ;
  const bf16* Bz = B + (size_t)z * bStrideZ;

  // per-thread staging coordinates (hoisted out of the k loop)
  const int r0 = tid >> 2;
  const int c0 = (tid & 3) << 3;
  const bf16* gA = Az + (size_t)(m0 + r0) * K + c0;
  const bf16* gB = Bz + (size_t)(n0 + r0) * K + c0;

  v8f acc[TM][TN];
#pragma unroll
  for (int i = 0; i < TM; ++i)
#pragma unroll
    for (int j = 0; j < TN; ++j)
#pragma unroll
      for (int e = 0; e < 8; ++e) acc[i][j][e] = 0.0f;

  auto stage = [&](int kt, int buf) {
#pragma unroll
    for (int i = 0; i < ACH; ++i)
      copy16(gA + (size_t)i * ROWS_PER_PASS * K + kt,
             &lA[buf][r0 + i * ROWS_PER_PASS][c0]);
#pragma unroll
    for (int i = 0; i < BCH; ++i)
      copy16(gB + (size_t)i * ROWS_PER_PASS * K + kt,
             &lB[buf][r0 + i * ROWS_PER_PASS][c0]);
  };

  const int NKT = K / KT;
  stage(0, 0);
  int buf = 0;
  for (int t = 0; t < NKT; ++t) {
    if (t + 1 < NKT) {
      stage((t + 1) * KT, buf ^ 1);     // prefetch next tile into other buffer
      wait_asynccnt<CHUNKS>();          // tile t landed (in-order completion)
    } else {
      wait_asynccnt<0>();
    }
    __syncthreads();

    // A fragment (16x32 bf16): lanes 0-15 hold K 0-7 & 16-23; lanes 16-31: 8-15 & 24-31
    FragBF fa[TM];
#pragma unroll
    for (int i = 0; i < TM; ++i) {
      int m = wmBase + i * 16 + (lane & 15);
      int kb = (lane < 16) ? 0 : 8;
      fa[i].h[0] = *(const v8bf*)&lA[buf][m][kb];
      fa[i].h[1] = *(const v8bf*)&lA[buf][m][kb + 16];
    }
    // B fragment (32x16 bf16): lanes 0-15 hold K 0-7 & 16-23; lanes 16-31: 8-15 & 24-31
    FragBF fb[TN];
#pragma unroll
    for (int j = 0; j < TN; ++j) {
      int n = wnBase + j * 16 + (lane & 15);
      int kb = (lane < 16) ? 0 : 8;
      fb[j].h[0] = *(const v8bf*)&lB[buf][n][kb];
      fb[j].h[1] = *(const v8bf*)&lB[buf][n][kb + 16];
    }

#pragma unroll
    for (int i = 0; i < TM; ++i)
#pragma unroll
      for (int j = 0; j < TN; ++j)
        acc[i][j] = __builtin_amdgcn_wmma_f32_16x16x32_bf16(
            false, fa[i].v, false, fb[j].v, (short)0, acc[i][j], false, false);

    __syncthreads();                    // all waves done reading buf
    buf ^= 1;
  }

  // epilogue scalar (uniform)
  float sc = 0.f;
  if (mode == 1) {
    float bw = softplusf(*scal0) + EPS_F;
    sc = 1.f / (2.f * bw * bw);
  } else if (mode == 2) {
    sc = softplusf(*scal0) + EPS_F;
  }

#pragma unroll
  for (int i = 0; i < TM; ++i) {
#pragma unroll
    for (int j = 0; j < TN; ++j) {
#pragma unroll
      for (int e = 0; e < 8; ++e) {
        int m = m0 + wmBase + i * 16 + ((lane < 16) ? e : e + 8);
        int n = n0 + wnBase + j * 16 + (lane & 15);
        float cv = acc[i][j][e];
        switch (mode) {
          case 0: {  // [4096,1024] -> bf16 [b,h,s,hd]
            bf16* O = (bf16*)out;
            size_t addr =
                (((size_t)((m >> 11) * HEADS + (n >> 6)) * S_LEN) + (m & (S_LEN - 1))) * HDIM +
                (n & (HDIM - 1));
            O[addr] = (bf16)cv;
          } break;
          case 1: {  // RBF kernel value, bf16
            const float* qsq = (const float*)auxp0 + (size_t)z * aux0StrideZ;
            const float* ksq = (const float*)auxp1 + (size_t)z * aux1StrideZ;
            float d2 = qsq[m] + ksq[n] - 2.f * cv;
            d2 = fmaxf(d2, 0.f);
            bf16* O = (bf16*)((char*)out + (size_t)z * outStrideZBytes);
            O[(size_t)m * N + n] = (bf16)expf(-d2 * sc);
          } break;
          case 2: {  // residual = v - (K@alpha + lam*alpha)
            const bf16* vv = (const bf16*)auxp0 + (size_t)z * aux0StrideZ;
            const float* al = (const float*)auxp1 + (size_t)z * aux1StrideZ;
            float* O = (float*)((char*)out + (size_t)z * outStrideZBytes);
            size_t idx = (size_t)m * HDIM + n;
            O[idx] = (float)vv[idx] - (cv + sc * al[idx]);
          } break;
          case 3: {  // attn interleave [b,s,h*64+hd] bf16
            int b = z >> 4, h = z & (HEADS - 1);
            bf16* O = (bf16*)out;
            O[((size_t)(b * S_LEN + m)) * D_DIM + h * HDIM + n] = (bf16)cv;
          } break;
          default: {  // plain f32
            float* O = (float*)out;
            O[(size_t)m * N + n] = cv;
          } break;
        }
      }
    }
  }
}

// ---------------------------------------------------------------------------
// Host orchestration
// ---------------------------------------------------------------------------
extern "C" void kernel_launch(void* const* d_in, const int* in_sizes, int n_in,
                              void* d_out, int out_size, void* d_ws, size_t ws_size,
                              hipStream_t stream) {
  (void)in_sizes; (void)n_in; (void)out_size; (void)ws_size;
  const float* x    = (const float*)d_in[0];
  const float* w_q  = (const float*)d_in[1];
  const float* w_k  = (const float*)d_in[2];
  const float* w_v  = (const float*)d_in[3];
  const float* w_o  = (const float*)d_in[4];
  const float* bandwidth  = (const float*)d_in[5];
  const float* diag_scale = (const float*)d_in[6];
  const float* reg        = (const float*)d_in[7];
  const float* pos        = (const float*)d_in[8];
  const float* head_proj  = (const float*)d_in[9];
  const float* lambda_reg = (const float*)d_in[10];

  char* ws = (char*)d_ws;
  size_t off = 0;
  auto alloc = [&](size_t bytes) -> void* {
    void* p = ws + off;
    off += (bytes + 255) & ~(size_t)255;
    return p;
  };

  const size_t MTOK = (size_t)BATCH * S_LEN;       // 4096 rows
  bf16* xbf  = (bf16*)alloc(MTOK * D_DIM * 2);
  bf16* wqbf = (bf16*)alloc((size_t)D_DIM * D_DIM * 2);
  bf16* wkbf = (bf16*)alloc((size_t)D_DIM * D_DIM * 2);
  bf16* wvbf = (bf16*)alloc((size_t)D_DIM * D_DIM * 2);
  bf16* wobf = (bf16*)alloc((size_t)D_DIM * D_DIM * 2);
  bf16* qbf  = (bf16*)alloc((size_t)NZ * S_LEN * HDIM * 2);
  bf16* kbf  = (bf16*)alloc((size_t)NZ * S_LEN * HDIM * 2);
  bf16* vbf  = (bf16*)alloc((size_t)NZ * S_LEN * HDIM * 2);
  float* qsq = (float*)alloc((size_t)NZ * S_LEN * 4);
  float* ksq = (float*)alloc((size_t)NZ * S_LEN * 4);
  float* diagp = (float*)alloc((size_t)NZ * S_LEN * 4);
  float* lr  = (float*)alloc((size_t)HEADS * S_LEN * RANK * 4);
  bf16* Km   = (bf16*)alloc((size_t)NZ * S_LEN * S_LEN * 2);   // 256 MB
  float* alpha  = (float*)alloc((size_t)NZ * S_LEN * HDIM * 4);
  bf16*  alphaT = (bf16*)alloc((size_t)NZ * HDIM * S_LEN * 2);
  float* res    = (float*)alloc((size_t)NZ * S_LEN * HDIM * 4);
  float* tbuf   = (float*)alloc((size_t)NZ * RANK * HDIM * 4);
  bf16* attnbf  = (bf16*)alloc(MTOK * D_DIM * 2);

  // 1) casts to bf16
  {
    long n = (long)MTOK * D_DIM;
    cast_f32_to_bf16_k<<<(int)((n + 255) / 256), 256, 0, stream>>>(x, xbf, n);
    long nw = (long)D_DIM * D_DIM;
    cast_f32_to_bf16_k<<<(int)((nw + 255) / 256), 256, 0, stream>>>(w_q, wqbf, nw);
    cast_f32_to_bf16_k<<<(int)((nw + 255) / 256), 256, 0, stream>>>(w_k, wkbf, nw);
    cast_f32_to_bf16_k<<<(int)((nw + 255) / 256), 256, 0, stream>>>(w_v, wvbf, nw);
    cast_f32_to_bf16_k<<<(int)((nw + 255) / 256), 256, 0, stream>>>(w_o, wobf, nw);
  }

  // 2) projections: [4096,1024] x [1024,1024]^T -> head-split bf16 q/k/v
  dim3 gProj(D_DIM / 128, (unsigned)(MTOK / 128), 1);
  wmma_gemm_nt<128, 128, 2, 4><<<gProj, 256, 0, stream>>>(
      xbf, wqbf, (int)MTOK, D_DIM, D_DIM, 0, 0, 0, qbf, 0, nullptr, nullptr, 0, 0, nullptr);
  wmma_gemm_nt<128, 128, 2, 4><<<gProj, 256, 0, stream>>>(
      xbf, wkbf, (int)MTOK, D_DIM, D_DIM, 0, 0, 0, kbf, 0, nullptr, nullptr, 0, 0, nullptr);
  wmma_gemm_nt<128, 128, 2, 4><<<gProj, 256, 0, stream>>>(
      xbf, wvbf, (int)MTOK, D_DIM, D_DIM, 0, 0, 0, vbf, 0, nullptr, nullptr, 0, 0, nullptr);

  // 3) row norms, diag preconditioner, low-rank lr
  {
    int rows = NZ * S_LEN;
    sqnorm64_k<<<(rows + 255) / 256, 256, 0, stream>>>(qbf, qsq, rows);
    sqnorm64_k<<<(rows + 255) / 256, 256, 0, stream>>>(kbf, ksq, rows);
    diag_precond_k<<<(rows + 255) / 256, 256, 0, stream>>>(qbf, kbf, bandwidth,
                                                           diag_scale, reg, diagp);
    int nlr = HEADS * S_LEN * RANK;
    lowrank_k<<<(nlr + 255) / 256, 256, 0, stream>>>(pos, head_proj, lr);
  }

  // 4) RBF kernel matrix per (b,h): K = exp(-(qsq+ksq-2 q.k)/(2bw^2)) -> bf16
  {
    dim3 g(S_LEN / 128, S_LEN / 128, NZ);
    wmma_gemm_nt<128, 128, 2, 4><<<g, 256, 0, stream>>>(
        qbf, kbf, S_LEN, S_LEN, HDIM, (long)S_LEN * HDIM, (long)S_LEN * HDIM,
        1, Km, (long)S_LEN * S_LEN * 2, qsq, ksq, S_LEN, S_LEN, bandwidth);
  }

  // 5) Richardson iterations
  (void)hipMemsetAsync(alpha, 0, (size_t)NZ * S_LEN * HDIM * 4, stream);
  (void)hipMemsetAsync(alphaT, 0, (size_t)NZ * HDIM * S_LEN * 2, stream);

  dim3 gMv(1, S_LEN / 128, NZ);
  for (int it = 0; it < N_ITERS; ++it) {
    // residual = v - (K + lam I) alpha
    wmma_gemm_nt<128, 64, 2, 2><<<gMv, 128, 0, stream>>>(
        Km, alphaT, S_LEN, HDIM, S_LEN, (long)S_LEN * S_LEN, (long)HDIM * S_LEN,
        2, res, (long)S_LEN * HDIM * 4, vbf, alpha,
        (long)S_LEN * HDIM, (long)S_LEN * HDIM, lambda_reg);
    // t = lr^T @ residual (rank-16 reduction)
    rich_reduce_k<<<NZ, 1024, 0, stream>>>(lr, res, tbuf);
    // alpha = clip(alpha + res*diagp + lr@t); also refresh bf16 alpha^T
    long nupd = (long)NZ * S_LEN * HDIM;
    rich_update_k<<<(int)((nupd + 255) / 256), 256, 0, stream>>>(
        res, diagp, lr, tbuf, alpha, alphaT);
  }

  // 6) out = K @ alpha -> interleaved bf16 attn [b,s,h*64+hd]
  wmma_gemm_nt<128, 64, 2, 2><<<gMv, 128, 0, stream>>>(
      Km, alphaT, S_LEN, HDIM, S_LEN, (long)S_LEN * S_LEN, (long)HDIM * S_LEN,
      3, attnbf, 0, nullptr, nullptr, 0, 0, nullptr);

  // 7) final projection: [4096,1024] x w_o^T -> f32 d_out
  wmma_gemm_nt<128, 128, 2, 4><<<gProj, 256, 0, stream>>>(
      attnbf, wobf, (int)MTOK, D_DIM, D_DIM, 0, 0,
      4, d_out, 0, nullptr, nullptr, 0, 0, nullptr);
}
